// Model_18923625906794
// MI455X (gfx1250) — matrix operand order repaired and checked by reading, not compiled, not run on hardware
//
#include <hip/hip_runtime.h>
#include <cmath>
#include <algorithm>

typedef __attribute__((ext_vector_type(16))) _Float16 v16h;
typedef __attribute__((ext_vector_type(4)))  _Float16 v4h;
typedef __attribute__((ext_vector_type(8)))  float    v8f;
typedef __attribute__((ext_vector_type(4)))  float    v4f;

#define DMODEL 512
#define NHEADS 8
#define DKH    64
#define NEGF  -1e30f
#define LDH    40   // LDS tile row stride in halves (pad vs bank conflicts)

// ---------------------------------------------------------------------------
// WMMA fragment loaders (16-bit, 16x16x32 f16 shapes)
// A 16x32 (MxK): lanes 0-15 -> M=lane, K=0..15; lanes 16-31 -> M=lane-16, K=16..31
// B 32x16 (KxN): lanes 0-15 -> N=lane, K=0..15; lanes 16-31 -> N=lane-16, K=16..31
// Tiles stored row-major [row][K] in LDS with row stride LDH halves.
// ---------------------------------------------------------------------------
__device__ inline v16h frag_a(const _Float16* tile, int lane) {
  int row = lane & 15;
  int kb  = (lane < 16) ? 0 : 16;
  const _Float16* p = tile + row * LDH + kb;
  v16h f;
#pragma unroll
  for (int i = 0; i < 8; ++i) f[i] = p[i];
#pragma unroll
  for (int i = 0; i < 8; ++i) f[8 + i] = p[8 + i];
  return f;
}

__device__ inline v16h frag_b(const _Float16* tile, int lane) {
  int col = lane & 15;
  int kb  = (lane < 16) ? 0 : 16;
  const _Float16* p = tile + col * LDH + kb;
  v16h f;
#pragma unroll
  for (int i = 0; i < 16; ++i) f[i] = p[i];
  return f;
}

// ---------------------------------------------------------------------------
// Tiled GEMM: C[M,N] = act( A[M,K] @ W[N,K]^T + bias + (accum ? C : 0) )
// REQUIRES: every tile full -> M % 128 == 0, N % 128 == 0, K % 32 == 0
// (holds for every GEMM in this model). 256 threads = 8 waves, BM=BN=128,
// BK=32; each wave computes 32x64 via 8 WMMAs. b128 global loads, b64 LDS
// stores, LDS double buffering: next tile's global loads are issued before
// the current tile's WMMAs so latency hides behind matrix math; one barrier
// per K-step. blockIdx.z batches with strides sA (A) / sC (C) in floats.
// ACT: 0 = none, 1 = exact GELU
// ---------------------------------------------------------------------------
template <int ACT>
__global__ __launch_bounds__(256) void gemm_wmma_k(
    const float* __restrict__ A, const float* __restrict__ W,
    const float* __restrict__ bias, float* __restrict__ C,
    int M, int N, int K, long sA, long sC, int accum) {
  __shared__ _Float16 As[2][128][LDH];
  __shared__ _Float16 Bs[2][128][LDH];
  int tid = threadIdx.x, lane = tid & 31, wid = tid >> 5;
  int wm = wid & 3, wn = wid >> 2;
  int bm = blockIdx.y * 128, bn = blockIdx.x * 128;
  const float* Ab = A + (long)blockIdx.z * sA;
  float*       Cb = C + (long)blockIdx.z * sC;

  v8f zero = {0.f, 0.f, 0.f, 0.f, 0.f, 0.f, 0.f, 0.f};
  v8f acc[2][4];
#pragma unroll
  for (int i = 0; i < 2; ++i)
#pragma unroll
    for (int j = 0; j < 4; ++j) acc[i][j] = zero;

  v4f ar[4], br[4];
  auto load_regs = [&](int k0) {
#pragma unroll
    for (int j = 0; j < 4; ++j) {
      int i = tid + j * 256;                 // 1024 float4 chunks per tile
      int r = i >> 3, c = (i & 7) * 4;
      ar[j] = *(const v4f*)(Ab + (long)(bm + r) * K + k0 + c);
      br[j] = *(const v4f*)(W  + (long)(bn + r) * K + k0 + c);
    }
  };
  auto store_lds = [&](int buf) {
#pragma unroll
    for (int j = 0; j < 4; ++j) {
      int i = tid + j * 256;
      int r = i >> 3, c = (i & 7) * 4;
      v4h ha, hb;
#pragma unroll
      for (int e = 0; e < 4; ++e) { ha[e] = (_Float16)ar[j][e]; hb[e] = (_Float16)br[j][e]; }
      *(v4h*)&As[buf][r][c] = ha;
      *(v4h*)&Bs[buf][r][c] = hb;
    }
  };

  load_regs(0);
  store_lds(0);
  __syncthreads();

  int cur = 0;
  for (int k0 = 0; k0 < K; k0 += 32) {
    bool next = (k0 + 32) < K;
    if (next) {
      load_regs(k0 + 32);                    // issue early; consumed after WMMAs
      if (k0 + 64 < K) {                     // gfx1250 global_prefetch_b8, tile k+2
        int r = tid >> 1, c = (tid & 1) * 16;
        __builtin_prefetch(Ab + (long)(bm + r) * K + k0 + 64 + c, 0, 0);
        __builtin_prefetch(W  + (long)(bn + r) * K + k0 + 64 + c, 0, 0);
      }
    }
    v16h a0 = frag_a(&As[cur][wm * 32][0], lane);
    v16h a1 = frag_a(&As[cur][wm * 32 + 16][0], lane);
#pragma unroll
    for (int j = 0; j < 4; ++j) {
      v16h bf = frag_b(&Bs[cur][wn * 64 + j * 16][0], lane);
      acc[0][j] = __builtin_amdgcn_wmma_f32_16x16x32_f16(false, a0, false, bf,
                                                         (short)0, acc[0][j], false, false);
      acc[1][j] = __builtin_amdgcn_wmma_f32_16x16x32_f16(false, a1, false, bf,
                                                         (short)0, acc[1][j], false, false);
    }
    if (next) {
      store_lds(cur ^ 1);                    // other buffer: no barrier needed before
      __syncthreads();                       // one barrier per K-step
    }
    cur ^= 1;
  }

  // C/D layout: VGPR j -> M = base + (lane<16?0:8) + j ; N = base + (lane&15)
  int lr = (lane < 16) ? 0 : 8, lc = lane & 15;
#pragma unroll
  for (int mi = 0; mi < 2; ++mi)
#pragma unroll
    for (int nj = 0; nj < 4; ++nj) {
      int gn = bn + wn * 64 + nj * 16 + lc;
      if (gn >= N) continue;
#pragma unroll
      for (int j = 0; j < 8; ++j) {
        int gm = bm + wm * 32 + mi * 16 + lr + j;
        if (gm >= M) continue;
        float v = acc[mi][nj][j];
        if (accum) v += Cb[(long)gm * N + gn];
        if (bias)  v += bias[gn];
        if (ACT == 1) v = 0.5f * v * (1.f + erff(v * 0.70710678f));
        Cb[(long)gm * N + gn] = v;
      }
    }
}

// ---------------------------------------------------------------------------
// series_decomp: moving average k=25 with edge clamp padding
// ---------------------------------------------------------------------------
__global__ void decomp_k(const float* __restrict__ x, float* seas, float* trend,
                         int B, int L, int C) {
  int i = blockIdx.x * blockDim.x + threadIdx.x;
  int tot = B * L * C;
  if (i >= tot) return;
  int c = i % C, l = (i / C) % L, b = i / (C * L);
  const float* xb = x + (long)b * L * C;
  float s = 0.f;
  for (int j = l - 12; j <= l + 12; ++j) {
    int jj = j < 0 ? 0 : (j >= L ? L - 1 : j);
    s += xb[jj * C + c];
  }
  float m = s * (1.f / 25.f);
  trend[i] = m;
  seas[i]  = x[i] - m;
}

// trend_out[b,p,c] = sum_l trend[b,l,c] * W[p,l] + tb[p]
__global__ void trendproj_k(const float* __restrict__ trend, const float* __restrict__ W,
                            const float* __restrict__ tb, float* out,
                            int B, int L, int P, int C) {
  int i = blockIdx.x * blockDim.x + threadIdx.x;
  int tot = B * P * C;
  if (i >= tot) return;
  int c = i % C, p = (i / C) % P, b = i / (C * P);
  const float* tr = trend + (long)b * L * C;
  const float* wr = W + (long)p * L;
  float s = tb[p];
  for (int l = 0; l < L; ++l) s += tr[l * C + c] * wr[l];
  out[i] = s;
}

// embed: circular conv1d(k=3) + time proj + sinusoidal positional embedding
__global__ void embed_k(const float* __restrict__ x, const float* __restrict__ mark,
                        const float* __restrict__ convw, const float* __restrict__ timew,
                        const float* __restrict__ timeb, float* out,
                        int B, int L, int Cin, int TFv) {
  int i = blockIdx.x * blockDim.x + threadIdx.x;
  int tot = B * L * DMODEL;
  if (i >= tot) return;
  int d = i % DMODEL, l = (i / DMODEL) % L, b = i / (DMODEL * L);
  float s = 0.f;
  for (int t = 0; t < 3; ++t) {
    int sl = l + t - 1;
    sl = sl < 0 ? L - 1 : (sl >= L ? 0 : sl);
    const float* xr = x + ((long)b * L + sl) * Cin;
    const float* wr = convw + ((long)t * DMODEL + d) * Cin;
    for (int c = 0; c < Cin; ++c) s += xr[c] * wr[c];
  }
  const float* mr = mark + ((long)b * L + l) * TFv;
  const float* tw = timew + (long)d * TFv;
  for (int f = 0; f < TFv; ++f) s += mr[f] * tw[f];
  s += timeb[d];
  float ang = (float)l * expf(-(float)(2 * (d >> 1)) * (9.2103404f / 512.f));
  s += (d & 1) ? cosf(ang) : sinf(ang);
  out[i] = s;
}

// circular pad: xp (B, L+2, D)
__global__ void padcirc_k(const float* __restrict__ x, float* xp, int B, int L, int D) {
  int i = blockIdx.x * blockDim.x + threadIdx.x;
  int tot = B * (L + 2) * D;
  if (i >= tot) return;
  int d = i % D, l = (i / D) % (L + 2), b = i / ((L + 2) * D);
  int sl = l - 1;
  sl = sl < 0 ? L - 1 : (sl >= L ? 0 : sl);
  xp[i] = x[((long)b * L + sl) * D + d];
}

// y = elu((y + cb) * bn_g/sqrt(1+eps) + bn_b)
__global__ void bnelu_k(float* y, const float* __restrict__ cb,
                        const float* __restrict__ bng, const float* __restrict__ bnb,
                        long n, int D) {
  long i = (long)blockIdx.x * blockDim.x + threadIdx.x;
  if (i >= n) return;
  int d = (int)(i % D);
  float v = y[i] + cb[d];
  v = v * (bng[d] * rsqrtf(1.f + 1e-5f)) + bnb[d];
  y[i] = v > 0.f ? v : (expf(v) - 1.f);
}

// maxpool k=3 stride=2, NEG padding: out[l'] = max(y[2l'-1..2l'+1])
__global__ void pool_k(const float* __restrict__ y, float* out, int B, int L, int D) {
  int Lo = L / 2;
  int i = blockIdx.x * blockDim.x + threadIdx.x;
  int tot = B * Lo * D;
  if (i >= tot) return;
  int d = i % D, lo = (i / D) % Lo, b = i / (D * Lo);
  float m = NEGF;
  for (int t = -1; t <= 1; ++t) {
    int l = 2 * lo + t;
    if (l >= 0 && l < L) m = fmaxf(m, y[((long)b * L + l) * D + d]);
  }
  out[i] = m;
}

// ProbSparse: sampled QK sparsity measure M[l] = max_u qk - sum_u qk / Lk
__global__ void samplem_k(const float* __restrict__ q, const float* __restrict__ k,
                          float* Mout, int H, int Lq, int Lk, int upart) {
  int bh = blockIdx.y;
  int l = blockIdx.x * blockDim.x + threadIdx.x;
  if (l >= Lq) return;
  int b = bh / H, h = bh % H;
  const float* qr = q + ((long)b * Lq + l) * DMODEL + h * DKH;
  float qv[DKH];
#pragma unroll
  for (int d = 0; d < DKH; ++d) qv[d] = qr[d];
  float mx = NEGF, sm = 0.f;
  for (int u = 0; u < upart; ++u) {
    unsigned idx = (2654435761u * (unsigned)(l * 89 + u * 127 + 13)) % (unsigned)Lk;
    const float* kr = k + ((long)b * Lk + idx) * DMODEL + h * DKH;
    float s = 0.f;
#pragma unroll
    for (int d = 0; d < DKH; ++d) s += qv[d] * kr[d];
    mx = fmaxf(mx, s);
    sm += s;
  }
  Mout[(long)bh * Lq + l] = mx - sm / (float)Lk;
}

// iterative top-k of M (descending); destroys M
__global__ void topk_k(float* M, int* top, int Lq, int ntop) {
  int bh = blockIdx.x, tid = threadIdx.x;
  float* Mr = M + (long)bh * Lq;
  __shared__ float sv[256];
  __shared__ int si[256];
  for (int it = 0; it < ntop; ++it) {
    float bv = -3e38f; int bi = 0;
    for (int l = tid; l < Lq; l += 256)
      if (Mr[l] > bv) { bv = Mr[l]; bi = l; }
    sv[tid] = bv; si[tid] = bi;
    __syncthreads();
    for (int off = 128; off > 0; off >>= 1) {
      if (tid < off && sv[tid + off] > sv[tid]) { sv[tid] = sv[tid + off]; si[tid] = si[tid + off]; }
      __syncthreads();
    }
    if (tid == 0) { top[bh * 64 + it] = si[0]; Mr[si[0]] = -3e38f; }
    __syncthreads();
  }
}

// context init: broadcast mean of V over keys (non-masked attention)
__global__ void ctxmean_k(const float* __restrict__ v, float* octx,
                          int B, int H, int Lq, int Lk) {
  int i = blockIdx.x * blockDim.x + threadIdx.x;
  int tot = B * H * DKH;
  if (i >= tot) return;
  int d = i % DKH, h = (i / DKH) % H, b = i / (DKH * H);
  float s = 0.f;
  for (int kk = 0; kk < Lk; ++kk) s += v[((long)b * Lk + kk) * DMODEL + h * DKH + d];
  s /= (float)Lk;
  for (int l = 0; l < Lq; ++l) octx[((long)b * Lq + l) * DMODEL + h * DKH + d] = s;
}

// context init: inclusive cumsum of V (masked self-attention, Lk==Lq)
__global__ void ctxcum_k(const float* __restrict__ v, float* octx,
                         int B, int H, int Lq, int Lk) {
  int i = blockIdx.x * blockDim.x + threadIdx.x;
  int tot = B * H * DKH;
  if (i >= tot) return;
  int d = i % DKH, h = (i / DKH) % H, b = i / (DKH * H);
  float s = 0.f;
  for (int l = 0; l < Lq; ++l) {
    s += v[((long)b * Lk + l) * DMODEL + h * DKH + d];
    octx[((long)b * Lq + l) * DMODEL + h * DKH + d] = s;
  }
}

// full attention for the n_top selected queries; scatter into context rows
__global__ void topattn_k(const float* __restrict__ q, const float* __restrict__ k,
                          const float* __restrict__ v, const int* __restrict__ top,
                          float* octx, int H, int Lq, int Lk, int ntop, int maskflag) {
  int u = blockIdx.x % ntop, bh = blockIdx.x / ntop;
  int b = bh / H, h = bh % H;
  int tl = top[bh * 64 + u];
  __shared__ float sc[1024];
  __shared__ float red[256];
  int tid = threadIdx.x;
  const float* qr = q + ((long)b * Lq + tl) * DMODEL + h * DKH;
  for (int kk = tid; kk < Lk; kk += 256) {
    const float* kr = k + ((long)b * Lk + kk) * DMODEL + h * DKH;
    float s = 0.f;
#pragma unroll
    for (int d = 0; d < DKH; ++d) s += qr[d] * kr[d];
    s *= 0.125f;  // 1/sqrt(64)
    if (maskflag && kk > tl) s = NEGF;
    sc[kk] = s;
  }
  __syncthreads();
  float m = NEGF;
  for (int kk = tid; kk < Lk; kk += 256) m = fmaxf(m, sc[kk]);
  red[tid] = m;
  __syncthreads();
  for (int off = 128; off > 0; off >>= 1) {
    if (tid < off) red[tid] = fmaxf(red[tid], red[tid + off]);
    __syncthreads();
  }
  m = red[0];
  __syncthreads();
  float s = 0.f;
  for (int kk = tid; kk < Lk; kk += 256) s += expf(sc[kk] - m);
  red[tid] = s;
  __syncthreads();
  for (int off = 128; off > 0; off >>= 1) {
    if (tid < off) red[tid] += red[tid + off];
    __syncthreads();
  }
  s = red[0];
  __syncthreads();
  if (tid < DKH) {
    float a = 0.f;
    for (int kk = 0; kk < Lk; ++kk)
      a += expf(sc[kk] - m) * v[((long)b * Lk + kk) * DMODEL + h * DKH + tid];
    octx[((long)b * Lq + tl) * DMODEL + h * DKH + tid] = a / s;
  }
}

// out = LayerNorm(x + delta) * g + b ; delta optional; one block per row
__global__ void addln_k(const float* __restrict__ x, const float* __restrict__ dlt,
                        const float* __restrict__ g, const float* __restrict__ bt,
                        float* out, int D) {
  int row = blockIdx.x, tid = threadIdx.x;
  const float* xr = x + (long)row * D;
  const float* dr = dlt ? dlt + (long)row * D : nullptr;
  __shared__ float r1[256], r2[256];
  float s = 0.f, s2 = 0.f;
  for (int i = tid; i < D; i += 256) {
    float v = xr[i] + (dr ? dr[i] : 0.f);
    s += v; s2 += v * v;
  }
  r1[tid] = s; r2[tid] = s2;
  __syncthreads();
  for (int off = 128; off > 0; off >>= 1) {
    if (tid < off) { r1[tid] += r1[tid + off]; r2[tid] += r2[tid + off]; }
    __syncthreads();
  }
  float mu = r1[0] / (float)D;
  float var = r2[0] / (float)D - mu * mu;
  float inv = rsqrtf(var + 1e-5f);
  __syncthreads();
  for (int i = tid; i < D; i += 256) {
    float v = xr[i] + (dr ? dr[i] : 0.f);
    out[(long)row * D + i] = (v - mu) * inv * g[i] + bt[i];
  }
}

// final projection (N=7) + trend add, on last PRED_LEN rows
__global__ void projout_k(const float* __restrict__ dec, const float* __restrict__ pw,
                          const float* __restrict__ pb, const float* __restrict__ trend,
                          float* out, int B, int P, int offRow, int Ldec) {
  int i = blockIdx.x * blockDim.x + threadIdx.x;
  int tot = B * P * 7;
  if (i >= tot) return;
  int c = i % 7, p = (i / 7) % P, b = i / (7 * P);
  const float* dr = dec + ((long)b * Ldec + offRow + p) * DMODEL;
  const float* wr = pw + (long)c * DMODEL;
  float s = pb[c];
  for (int d = 0; d < DMODEL; ++d) s += dr[d] * wr[d];
  out[i] = s + trend[i];
}

// ---------------------------------------------------------------------------
// Host orchestration
// ---------------------------------------------------------------------------
struct AttnP { const float *bk, *bo, *bq, *bv, *wk, *wo, *wq, *wv; };
struct FfnP  { const float *b1, *b2, *w1, *w2; };
struct LnP   { const float *b, *g; };
struct ConvP { const float *b, *bn_b, *bn_g, *w; };

extern "C" void kernel_launch(void* const* d_in, const int* in_sizes, int n_in,
                              void* d_out, int out_size, void* d_ws, size_t ws_size,
                              hipStream_t stream) {
  (void)in_sizes; (void)n_in; (void)out_size; (void)ws_size;
  const int B = 8, SL = 1024, PL = 336, CIN = 7, TF = 4, DECL = 848;

  // ---- inputs ----
  const float* x_enc      = (const float*)d_in[0];
  const float* x_mark_enc = (const float*)d_in[1];
  const float* x_dec      = (const float*)d_in[2];
  const float* x_mark_dec = (const float*)d_in[3];

  // ---- params: pytree flatten order (sorted dict keys, lists in order) ----
  int ip = 4;
  auto nx = [&]() -> const float* { return (const float*)d_in[ip++]; };
  auto rdAttn = [&](AttnP& a) { a.bk=nx(); a.bo=nx(); a.bq=nx(); a.bv=nx();
                                a.wk=nx(); a.wo=nx(); a.wq=nx(); a.wv=nx(); };
  auto rdFfn  = [&](FfnP& f)  { f.b1=nx(); f.b2=nx(); f.w1=nx(); f.w2=nx(); };
  auto rdLn   = [&](LnP& l)   { l.b=nx(); l.g=nx(); };

  const float *dec_conv_w = nx(), *dec_time_b = nx(), *dec_time_w = nx();
  AttnP dcross[2], dself[2]; FfnP dffn[2]; LnP dln1[2], dln2[2], dln3[2];
  for (int i = 0; i < 2; ++i) {
    rdAttn(dcross[i]); rdFfn(dffn[i]);
    rdLn(dln1[i]); rdLn(dln2[i]); rdLn(dln3[i]);
    rdAttn(dself[i]);
  }
  LnP dec_norm; rdLn(dec_norm);
  ConvP econv[2];
  for (int i = 0; i < 2; ++i) { econv[i].b=nx(); econv[i].bn_b=nx(); econv[i].bn_g=nx(); econv[i].w=nx(); }
  const float *enc_conv_w = nx(), *enc_time_b = nx(), *enc_time_w = nx();
  AttnP eattn[3]; FfnP effn[3]; LnP eln1[3], eln2[3];
  for (int i = 0; i < 3; ++i) { rdAttn(eattn[i]); rdFfn(effn[i]); rdLn(eln1[i]); rdLn(eln2[i]); }
  LnP enc_norm; rdLn(enc_norm);
  const float *proj_b = nx(), *proj_w = nx(), *trend_b = nx(), *trend_w = nx();

  // ---- workspace carve ----
  float* w = (float*)d_ws;
  size_t off = 0;
  auto alloc = [&](size_t n) { float* r = w + off; off += n; return r; };
  const size_t ACT = (size_t)8192 * DMODEL;          // max B*L rows * 512
  float* region = alloc(4 * ACT);                    // qb|kb|vb|octx == ffnh == xpad
  float* qb = region, *kb = region + ACT, *vb = region + 2 * ACT, *octx = region + 3 * ACT;
  float* ffnh = region;                              // 8192 x 2048
  float* xpad = region;                              // B x (L+2) x 512
  float* xe   = alloc(ACT);                          // encoder activation
  float* xd   = alloc((size_t)B * DECL * DMODEL);    // decoder activation
  float* tmp  = alloc(ACT);                          // GEMM outputs / conv
  float* seas = alloc((size_t)B * SL * CIN);
  float* trnd = alloc((size_t)B * SL * CIN);
  float* tout = alloc((size_t)B * PL * CIN);
  float* Mbuf = alloc((size_t)B * NHEADS * SL);
  int*   topb = (int*)alloc((size_t)B * NHEADS * 64);

  auto gemm = [&](const float* A, const float* W_, const float* bias, float* C,
                  int M, int N, int K, int act, int accum,
                  int batch = 1, long sA = 0, long sC = 0) {
    dim3 g((N + 127) / 128, (M + 127) / 128, batch);
    if (act == 1) gemm_wmma_k<1><<<g, 256, 0, stream>>>(A, W_, bias, C, M, N, K, sA, sC, accum);
    else          gemm_wmma_k<0><<<g, 256, 0, stream>>>(A, W_, bias, C, M, N, K, sA, sC, accum);
  };
  auto blk = [](long n) { return (unsigned)((n + 255) / 256); };

  auto run_attn = [&](float* x, const float* xkv, int Lq, int Lk,
                      const AttnP& a, const LnP& ln, int mask) {
    int Mq = B * Lq, Mk = B * Lk;
    gemm(x,   a.wq, a.bq, qb, Mq, DMODEL, DMODEL, 0, 0);
    gemm(xkv, a.wk, a.bk, kb, Mk, DMODEL, DMODEL, 0, 0);
    gemm(xkv, a.wv, a.bv, vb, Mk, DMODEL, DMODEL, 0, 0);
    int up = std::min(5 * (int)ceil(log((double)Lk)), Lk);
    int nt = std::min(5 * (int)ceil(log((double)Lq)), Lq);
    dim3 gs((Lq + 127) / 128, B * NHEADS);
    samplem_k<<<gs, 128, 0, stream>>>(qb, kb, Mbuf, NHEADS, Lq, Lk, up);
    topk_k<<<B * NHEADS, 256, 0, stream>>>(Mbuf, topb, Lq, nt);
    if (mask) ctxcum_k<<<blk(B * NHEADS * DKH), 256, 0, stream>>>(vb, octx, B, NHEADS, Lq, Lk);
    else      ctxmean_k<<<blk(B * NHEADS * DKH), 256, 0, stream>>>(vb, octx, B, NHEADS, Lq, Lk);
    topattn_k<<<B * NHEADS * nt, 256, 0, stream>>>(qb, kb, vb, topb, octx, NHEADS, Lq, Lk, nt, mask);
    gemm(octx, a.wo, a.bo, tmp, Mq, DMODEL, DMODEL, 0, 0);
    addln_k<<<Mq, 256, 0, stream>>>(x, tmp, ln.g, ln.b, x, DMODEL);
  };
  auto run_ffn = [&](float* x, int L, const FfnP& f, const LnP& ln) {
    int M = B * L;
    gemm(x,    f.w1, f.b1, ffnh, M, 2048, DMODEL, 1, 0);   // GELU fused
    gemm(ffnh, f.w2, f.b2, tmp,  M, DMODEL, 2048, 0, 0);
    addln_k<<<M, 256, 0, stream>>>(x, tmp, ln.g, ln.b, x, DMODEL);
  };

  // ---- series decomposition + trend projection ----
  decomp_k<<<blk((long)B * SL * CIN), 256, 0, stream>>>(x_enc, seas, trnd, B, SL, CIN);
  trendproj_k<<<blk((long)B * PL * CIN), 256, 0, stream>>>(trnd, trend_w, trend_b, tout, B, SL, PL, CIN);

  // ---- encoder ----
  embed_k<<<blk((long)B * SL * DMODEL), 256, 0, stream>>>(seas, x_mark_enc, enc_conv_w,
                                                          enc_time_w, enc_time_b, xe, B, SL, CIN, TF);
  int L = SL;
  for (int il = 0; il < 3; ++il) {
    run_attn(xe, xe, L, L, eattn[il], eln1[il], 0);
    run_ffn(xe, L, effn[il], eln2[il]);
    if (il < 2) {  // conv distil: circular conv(k=3) as 3 shifted batched GEMMs
      padcirc_k<<<blk((long)B * (L + 2) * DMODEL), 256, 0, stream>>>(xe, xpad, B, L, DMODEL);
      for (int t = 0; t < 3; ++t)
        gemm(xpad + (long)t * DMODEL, econv[il].w + (long)t * DMODEL * DMODEL, nullptr, tmp,
             L, DMODEL, DMODEL, 0, t > 0, B, (long)(L + 2) * DMODEL, (long)L * DMODEL);
      bnelu_k<<<blk((long)B * L * DMODEL), 256, 0, stream>>>(tmp, econv[il].b, econv[il].bn_g,
                                                             econv[il].bn_b, (long)B * L * DMODEL, DMODEL);
      pool_k<<<blk((long)B * (L / 2) * DMODEL), 256, 0, stream>>>(tmp, xe, B, L, DMODEL);
      L /= 2;
    }
  }
  addln_k<<<B * L, 256, 0, stream>>>(xe, nullptr, enc_norm.g, enc_norm.b, xe, DMODEL);

  // ---- decoder ----
  embed_k<<<blk((long)B * DECL * DMODEL), 256, 0, stream>>>(x_dec, x_mark_dec, dec_conv_w,
                                                            dec_time_w, dec_time_b, xd, B, DECL, CIN, TF);
  for (int il = 0; il < 2; ++il) {
    run_attn(xd, xd, DECL, DECL, dself[il], dln1[il], 1);   // masked self
    run_attn(xd, xe, DECL, L,    dcross[il], dln2[il], 0);  // cross with encoder (L=256)
    run_ffn(xd, DECL, dffn[il], dln3[il]);
  }
  addln_k<<<B * DECL, 256, 0, stream>>>(xd, nullptr, dec_norm.g, dec_norm.b, xd, DMODEL);

  // ---- output projection + trend ----
  projout_k<<<blk((long)B * PL * 7), 256, 0, stream>>>(xd, proj_w, proj_b, tout,
                                                       (float*)d_out, B, PL, DECL - PL, DECL);
}
